// PriorityComputation_13623636263379
// MI455X (gfx1250) — compile-verified, matrix operand order checked by reading, not run on hardware
//
#include <hip/hip_runtime.h>
#include <hip/hip_bf16.h>

#define B_SEG  64
#define LATENT 256

typedef __attribute__((ext_vector_type(2)))  float  v2f;
typedef __attribute__((ext_vector_type(8)))  float  v8f;
typedef __attribute__((ext_vector_type(16))) __bf16 v16bf;

// Monotone map float -> uint32 so unsigned atomicMax implements float max
// (handles negatives; -0 < +0; init value 0 is below ord(-inf)=0x007FFFFF).
__device__ __forceinline__ unsigned f2ord(float f) {
  unsigned u = __float_as_uint(f);
  return (u & 0x80000000u) ? ~u : (u | 0x80000000u);
}
__device__ __forceinline__ float ord2f(unsigned u) {
  u = (u & 0x80000000u) ? (u & 0x7FFFFFFFu) : ~u;
  return __uint_as_float(u);
}

// ---------------------------------------------------------------------------
// Workspace init: seg-max (ordered uints) = 0, seg-sum = 0.0f. Runs every call.
// ---------------------------------------------------------------------------
__global__ void ws_init_kernel(unsigned* __restrict__ gmax, float* __restrict__ gsum) {
  int i = threadIdx.x;
  if (i < B_SEG) { gmax[i] = 0u; gsum[i] = 0.0f; }
}

// ---------------------------------------------------------------------------
// uncertainty[b] = sum_k 0.5*log(2*pi*e*std^2), computed as (entropy) x (ones)
// with WMMA. 1 block, 128 threads = 4 wave32; wave w owns rows 16w..16w+15.
// ---------------------------------------------------------------------------
__global__ void unc_wmma_kernel(const float* __restrict__ std_, float* __restrict__ unc) {
  const int lane = threadIdx.x & 31;
  const int wave = threadIdx.x >> 5;
  const int row  = wave * 16 + (lane & 15);
  const int half = lane >> 4;                 // 0: lanes 0-15, 1: lanes 16-31
  const float TPE = 17.0794684453471341f;     // 2*pi*e
  v8f acc = {};

#if __has_builtin(__builtin_amdgcn_wmma_f32_16x16x4_f32)
  // f32 A 16x4 layout: lanes 0-15 hold K={0,1}, lanes 16-31 hold K={2,3}.
  v2f ones; ones.x = 1.0f; ones.y = 1.0f;
  const float* rp = std_ + (size_t)row * LATENT + 2 * half;
  for (int kb = 0; kb < LATENT; kb += 4) {
    float s0 = rp[kb];
    float s1 = rp[kb + 1];
    v2f a;
    a.x = 0.5f * __logf(TPE * s0 * s0);
    a.y = 0.5f * __logf(TPE * s1 * s1);
    acc = __builtin_amdgcn_wmma_f32_16x16x4_f32(false, a, false, ones,
                                                (short)0, acc, false, false);
  }
#else
  // Fallback: codegen-confirmed bf16 WMMA 16x16x32.
  // 16-bit A 16x32 layout: slot v (of 8 value-pairs) holds
  // K = (v<4 ? 0 : 16) + 8*half + 2*(v&3) + {0,1}.
  v16bf ones;
  #pragma unroll
  for (int i = 0; i < 16; ++i) ones[i] = (__bf16)1.0f;
  for (int kb = 0; kb < LATENT; kb += 32) {
    v16bf a;
    #pragma unroll
    for (int v = 0; v < 8; ++v) {
      int k = ((v & 4) ? 16 : 0) + 8 * half + 2 * (v & 3);
      float s0 = std_[(size_t)row * LATENT + kb + k];
      float s1 = std_[(size_t)row * LATENT + kb + k + 1];
      a[2 * v]     = (__bf16)(0.5f * __logf(TPE * s0 * s0));
      a[2 * v + 1] = (__bf16)(0.5f * __logf(TPE * s1 * s1));
    }
    acc = __builtin_amdgcn_wmma_f32_16x16x32_bf16(false, a, false, ones,
                                                  (short)0, acc, false, false);
  }
#endif

  // Every column of D equals the row sums; read column N=0.
  // C/D layout: lane 0 holds M=0..7 (VGPR 0..7), lane 16 holds M=8..15.
  if ((lane & 15) == 0) {
    const int base = wave * 16 + half * 8;
    #pragma unroll
    for (int r = 0; r < 8; ++r) unc[base + r] = acc[r];
  }
}

// ---------------------------------------------------------------------------
// Pass 1: priority[i] = coh[i] * unc[batch[i]]; segment max via LDS bins.
// ---------------------------------------------------------------------------
__global__ void pass1_kernel(const float* __restrict__ coh, const int* __restrict__ batch,
                             const float* __restrict__ unc, float* __restrict__ prio,
                             unsigned* __restrict__ gmax, int n) {
  __shared__ float    s_unc[B_SEG];
  __shared__ unsigned s_max[B_SEG];
  for (int i = threadIdx.x; i < B_SEG; i += blockDim.x) { s_unc[i] = unc[i]; s_max[i] = 0u; }
  __syncthreads();

  const int tid    = blockIdx.x * blockDim.x + threadIdx.x;
  const int stride = gridDim.x * blockDim.x;
  const int n4     = n >> 2;
  const float4* c4 = (const float4*)coh;
  const int4*   b4 = (const int4*)batch;
  float4*       p4 = (float4*)prio;

  for (int i = tid; i < n4; i += stride) {
    float4 c = c4[i];
    int4   b = b4[i];
    float4 p;
    p.x = c.x * s_unc[b.x]; p.y = c.y * s_unc[b.y];
    p.z = c.z * s_unc[b.z]; p.w = c.w * s_unc[b.w];
    p4[i] = p;
    atomicMax(&s_max[b.x], f2ord(p.x));
    atomicMax(&s_max[b.y], f2ord(p.y));
    atomicMax(&s_max[b.z], f2ord(p.z));
    atomicMax(&s_max[b.w], f2ord(p.w));
  }
  for (int i = (n4 << 2) + tid; i < n; i += stride) {   // tail
    int b = batch[i];
    float p = coh[i] * s_unc[b];
    prio[i] = p;
    atomicMax(&s_max[b], f2ord(p));
  }
  __syncthreads();
  for (int i = threadIdx.x; i < B_SEG; i += blockDim.x) atomicMax(&gmax[i], s_max[i]);
}

// ---------------------------------------------------------------------------
// Pass 2: e[i] = exp(prio[i] - segmax[batch[i]]); segment sum via LDS bins.
// ---------------------------------------------------------------------------
__global__ void pass2_kernel(const float* __restrict__ prio, const int* __restrict__ batch,
                             const unsigned* __restrict__ gmax, float* __restrict__ evals,
                             float* __restrict__ gsum, int n) {
  __shared__ float s_max[B_SEG];
  __shared__ float s_sum[B_SEG];
  for (int i = threadIdx.x; i < B_SEG; i += blockDim.x) { s_max[i] = ord2f(gmax[i]); s_sum[i] = 0.0f; }
  __syncthreads();

  const int tid    = blockIdx.x * blockDim.x + threadIdx.x;
  const int stride = gridDim.x * blockDim.x;
  const int n4     = n >> 2;
  const float4* p4 = (const float4*)prio;
  const int4*   b4 = (const int4*)batch;
  float4*       e4 = (float4*)evals;

  for (int i = tid; i < n4; i += stride) {
    float4 p = p4[i];
    int4   b = b4[i];
    float4 e;
    e.x = __expf(p.x - s_max[b.x]); e.y = __expf(p.y - s_max[b.y]);
    e.z = __expf(p.z - s_max[b.z]); e.w = __expf(p.w - s_max[b.w]);
    e4[i] = e;
    atomicAdd(&s_sum[b.x], e.x);
    atomicAdd(&s_sum[b.y], e.y);
    atomicAdd(&s_sum[b.z], e.z);
    atomicAdd(&s_sum[b.w], e.w);
  }
  for (int i = (n4 << 2) + tid; i < n; i += stride) {   // tail
    int b = batch[i];
    float e = __expf(prio[i] - s_max[b]);
    evals[i] = e;
    atomicAdd(&s_sum[b], e);
  }
  __syncthreads();
  for (int i = threadIdx.x; i < B_SEG; i += blockDim.x) atomicAdd(&gsum[i], s_sum[i]);
}

// ---------------------------------------------------------------------------
// Pass 3: normalized[i] = e[i] / segsum[batch[i]]  (in place over e buffer).
// ---------------------------------------------------------------------------
__global__ void pass3_kernel(const int* __restrict__ batch, const float* __restrict__ gsum,
                             float* __restrict__ enorm, int n) {
  __shared__ float s_inv[B_SEG];
  for (int i = threadIdx.x; i < B_SEG; i += blockDim.x) s_inv[i] = __frcp_rn(gsum[i]);
  __syncthreads();

  const int tid    = blockIdx.x * blockDim.x + threadIdx.x;
  const int stride = gridDim.x * blockDim.x;
  const int n4     = n >> 2;
  const int4* b4   = (const int4*)batch;
  float4*     e4   = (float4*)enorm;

  for (int i = tid; i < n4; i += stride) {
    float4 e = e4[i];
    int4   b = b4[i];
    e.x *= s_inv[b.x]; e.y *= s_inv[b.y];
    e.z *= s_inv[b.z]; e.w *= s_inv[b.w];
    e4[i] = e;
  }
  for (int i = (n4 << 2) + tid; i < n; i += stride)     // tail
    enorm[i] *= s_inv[batch[i]];
}

// ---------------------------------------------------------------------------
extern "C" void kernel_launch(void* const* d_in, const int* in_sizes, int n_in,
                              void* d_out, int out_size, void* d_ws, size_t ws_size,
                              hipStream_t stream) {
  const float* coh   = (const float*)d_in[0];  // (N,)
  // d_in[1] = posterior_mean (unused by the reference math)
  const float* pstd  = (const float*)d_in[2];  // (B, LATENT)
  const int*   batch = (const int*)d_in[3];    // (N,)
  const int n = in_sizes[0];

  float* out   = (float*)d_out;
  float* prio  = out;                          // output 0: priority        (N)
  float* pnorm = out + (size_t)n;              // output 1: normalized      (N)
  float* unc   = out + 2 * (size_t)n;          // output 2: uncertainty     (B)

  unsigned* gmax = (unsigned*)d_ws;            // 64 ordered-uint seg maxes
  float*    gsum = (float*)d_ws + B_SEG;       // 64 float seg sums

  ws_init_kernel<<<1, 64, 0, stream>>>(gmax, gsum);
  unc_wmma_kernel<<<1, 128, 0, stream>>>(pstd, unc);

  const int blocks = 2048, threads = 256;
  pass1_kernel<<<blocks, threads, 0, stream>>>(coh, batch, unc, prio, gmax, n);
  pass2_kernel<<<blocks, threads, 0, stream>>>(prio, batch, gmax, pnorm, gsum, n);
  pass3_kernel<<<blocks, threads, 0, stream>>>(batch, gsum, pnorm, n);
}
